// MultiHeadedSelfAttention_73263552135571
// MI455X (gfx1250) — compile-verified, matrix operand order checked by reading, not run on hardware
//
#include <hip/hip_runtime.h>

// ---------- CDNA5 WMMA types ----------
typedef __attribute__((ext_vector_type(16))) __bf16 v16bf;
typedef __attribute__((ext_vector_type(8)))  __bf16 v8bf;
typedef __attribute__((ext_vector_type(4)))  __bf16 v4bf;
typedef __attribute__((ext_vector_type(8)))  float  v8f;
typedef __attribute__((ext_vector_type(4)))  float  v4f;

// ---------- TDM descriptor vector types ----------
typedef __attribute__((ext_vector_type(4))) unsigned int u32x4;
typedef __attribute__((ext_vector_type(8))) int          i32x8;
typedef __attribute__((ext_vector_type(4))) int          i32x4;

union UB16 { v16bf v; v8bf h[2]; };
union UF8  { v8f v;  float f[8]; };

__device__ __forceinline__ v8f wmma_bf16(v16bf a, v16bf b, v8f c) {
  // D(16x16 f32) = A(16x32 bf16) * B(32x16 bf16) + C
  return __builtin_amdgcn_wmma_f32_16x16x32_bf16(
      /*neg_a=*/false, a, /*neg_b=*/false, b,
      /*c_mod=*/(short)0, c, /*reuse_a=*/false, /*reuse_b=*/false);
}

static constexpr int Bb  = 2;
static constexpr int Ss  = 2048;
static constexpr int Dd  = 1024;
static constexpr int Hh  = 16;
static constexpr int DKk = 64;

// Issue a TDM 2D tile load: tile_dim0 = 32 elems (64B rows), tile_dim1 = 128 rows,
// bf16 data, row stride = K elements in global, LDS rows padded 64B -> 80B
// (pad_interval code 3 = 16 DWORDs, pad_amount code 3 = 4 DWORDs).
__device__ __forceinline__ void tdm_load_tile_A(unsigned lds_off,
                                                const __bf16* gptr,
                                                int M, int K) {
  unsigned long long ga = (unsigned long long)(uintptr_t)gptr;
  u32x4 g0;
  g0[0] = 1u;                                             // count=1, user mode
  g0[1] = lds_off;                                        // lds_addr
  g0[2] = (unsigned)(ga & 0xFFFFFFFFu);                   // global_addr[31:0]
  g0[3] = (unsigned)((ga >> 32) & 0x1FFFFFFu) | (2u << 30); // addr[56:32] | type=2

  i32x8 g1;
  g1[0] = (int)((1u << 16)    // data_size = 1 (2 bytes)
              | (1u << 20)    // pad_enable
              | (3u << 22)    // pad_interval: 16 DWORDs (one 64B row)
              | (3u << 25));  // pad_amount: 4 DWORDs (16B pad -> 80B stride)
  g1[1] = (int)(((unsigned)K & 0xFFFFu) << 16);                       // tensor_dim0[15:0]
  g1[2] = (int)((((unsigned)K >> 16) & 0xFFFFu) |
                (((unsigned)M & 0xFFFFu) << 16));                     // dim0[31:16] | dim1[15:0]
  g1[3] = (int)((((unsigned)M >> 16) & 0xFFFFu) | (32u << 16));       // dim1[31:16] | tile_dim0=32
  g1[4] = 128;                                                        // tile_dim1=128, tile_dim2=0
  g1[5] = K;                                                          // tensor_dim0_stride[31:0]
  g1[6] = 0;
  g1[7] = 0;

  i32x4 g2 = {0, 0, 0, 0};
  i32x4 g3 = {0, 0, 0, 0};
#if defined(__clang_major__) && (__clang_major__ >= 23)
  i32x8 g4 = {0, 0, 0, 0, 0, 0, 0, 0};
  __builtin_amdgcn_tensor_load_to_lds(g0, g1, g2, g3, g4, 0);
#else
  __builtin_amdgcn_tensor_load_to_lds(g0, g1, g2, g3, 0);
#endif
}

// ---------- elementwise prep ----------
__global__ __launch_bounds__(256)
void add_pe_cvt(const float* __restrict__ q, const float* __restrict__ pe,
                __bf16* __restrict__ x, int total, int SD) {
  int i = (blockIdx.x * blockDim.x + threadIdx.x) * 4;
  if (i >= total) return;
  v4f a = *(const v4f*)(q + i);
  v4f b = *(const v4f*)(pe + (i % SD));
  v4bf o;
  o[0] = (__bf16)(a[0] + b[0]);
  o[1] = (__bf16)(a[1] + b[1]);
  o[2] = (__bf16)(a[2] + b[2]);
  o[3] = (__bf16)(a[3] + b[3]);
  *(v4bf*)(x + i) = o;
}

__global__ __launch_bounds__(256)
void cvt_bf16(const float* __restrict__ in, __bf16* __restrict__ out, int total) {
  int i = (blockIdx.x * blockDim.x + threadIdx.x) * 4;
  if (i >= total) return;
  v4f a = *(const v4f*)(in + i);
  v4bf o;
  o[0] = (__bf16)a[0]; o[1] = (__bf16)a[1];
  o[2] = (__bf16)a[2]; o[3] = (__bf16)a[3];
  *(v4bf*)(out + i) = o;
}

// ---------- tiled WMMA GEMM: C(MxN) = A(MxK,bf16) * B(KxN,bf16) ----------
// A tile staged by the Tensor Data Mover (global -> LDS, padded rows);
// B tile staged manually with a transpose into Bs[n][k].
// MODE 0: write f32 to Cf row-major.
// MODE 1: scatter into q (scaled 1/8), k [(b,h,s,dk)] and vT [(b,h,dk,s)].
template<int MODE>
__global__ __launch_bounds__(256)
void gemm_bf16_128x128(const __bf16* __restrict__ A, const __bf16* __restrict__ Bw,
                       float* __restrict__ Cf, int M, int N, int K,
                       __bf16* __restrict__ qo, __bf16* __restrict__ ko,
                       __bf16* __restrict__ vto) {
  __shared__ alignas(16) __bf16 As[128][40];   // [m][k], row stride 80B (16B-aligned)
  __shared__ alignas(16) __bf16 Bs[128][40];   // [n][k] (transposed)

  const int tid    = threadIdx.x;
  const int l      = tid & 31;
  const int w      = tid >> 5;        // 8 waves
  const int lane16 = l & 15;
  const int hh     = l >> 4;          // half-wave select
  const int wr     = w & 1;           // 2 wave-rows of 64
  const int wc     = w >> 1;          // 4 wave-cols of 32
  const int mBase  = blockIdx.y * 128;
  const int nBase  = blockIdx.x * 128;
  const unsigned lds_as = (unsigned)(uintptr_t)&As[0][0];

  UF8 acc[4][2];
  #pragma unroll
  for (int a = 0; a < 4; a++)
    #pragma unroll
    for (int b = 0; b < 2; b++) {
      v8f z = {0.f,0.f,0.f,0.f,0.f,0.f,0.f,0.f};
      acc[a][b].v = z;
    }

  const int bk = tid >> 3, bp = tid & 7;   // B stage: 32 k-rows x 8 col-chunks of 16

  for (int k0 = 0; k0 < K; k0 += 32) {
    // ---- stage A tile (128x32) via Tensor Data Mover ----
    if (w == 0) {
      tdm_load_tile_A(lds_as, A + (size_t)mBase * K + k0, M, K);
    }
    // ---- stage B tile (32x128), transposed into Bs[n][k] ----
    const __bf16* bg = Bw + (size_t)(k0 + bk) * N + nBase + bp * 16;
    v8bf b0 = *(const v8bf*)bg;
    v8bf b1 = *(const v8bf*)(bg + 8);
    #pragma unroll
    for (int i = 0; i < 8; i++) {
      Bs[bp * 16 + i][bk]     = b0[i];
      Bs[bp * 16 + 8 + i][bk] = b1[i];
    }
    if (k0 + 32 < K) {
      __builtin_prefetch(bg + (size_t)32 * N, 0, 1);     // global_prefetch_b8
    }
    if (w == 0) {
      __builtin_amdgcn_s_wait_tensorcnt(0);              // TDM tile landed in LDS
    }
    __syncthreads();

    // ---- fragments ----
    UB16 af[4], bfr[2];
    #pragma unroll
    for (int mt = 0; mt < 4; mt++) {
      int m = wr * 64 + mt * 16 + lane16;
      af[mt].h[0] = *(const v8bf*)&As[m][hh * 8];
      af[mt].h[1] = *(const v8bf*)&As[m][16 + hh * 8];
    }
    #pragma unroll
    for (int nf = 0; nf < 2; nf++) {
      int n = wc * 32 + nf * 16 + lane16;
      bfr[nf].h[0] = *(const v8bf*)&Bs[n][hh * 8];
      bfr[nf].h[1] = *(const v8bf*)&Bs[n][16 + hh * 8];
    }
    #pragma unroll
    for (int mt = 0; mt < 4; mt++)
      #pragma unroll
      for (int nf = 0; nf < 2; nf++)
        acc[mt][nf].v = wmma_bf16(af[mt].v, bfr[nf].v, acc[mt][nf].v);
    __syncthreads();
  }

  // ---- epilogue ----
  #pragma unroll
  for (int mt = 0; mt < 4; mt++) {
    #pragma unroll
    for (int nf = 0; nf < 2; nf++) {
      #pragma unroll
      for (int v = 0; v < 8; v++) {
        int row = mBase + wr * 64 + mt * 16 + v + 8 * hh;
        int col = nBase + wc * 32 + nf * 16 + lane16;
        float c = acc[mt][nf].f[v];
        if (MODE == 0) {
          Cf[(size_t)row * N + col] = c;
        } else {
          int b    = row >> 11;          // row / 2048
          int s    = row & 2047;
          int head = col / 192;          // 3*dk per head
          int rem  = col - head * 192;
          int t    = rem >> 6;
          int dd   = rem & 63;
          size_t idx = ((size_t)(b * Hh + head) * Ss + s) * DKk + dd;
          if (t == 0)      qo[idx] = (__bf16)(c * 0.125f);   // fold d_k^-0.5
          else if (t == 1) ko[idx] = (__bf16)c;
          else             vto[((size_t)(b * Hh + head) * DKk + dd) * Ss + s] = (__bf16)c;
        }
      }
    }
  }
}

// ---------- flash attention: one wave = 16 query rows, stream 32 keys/iter ----------
__global__ __launch_bounds__(128)
void flash_attn(const __bf16* __restrict__ q, const __bf16* __restrict__ k,
                const __bf16* __restrict__ vt, __bf16* __restrict__ o) {
  __shared__ alignas(16) __bf16 Ps[4][16][40];   // per-wave P tile (16x32)
  const int tid    = threadIdx.x;
  const int l      = tid & 31;
  const int w      = tid >> 5;
  const int lane16 = l & 15;
  const int hh     = l >> 4;
  const int bh     = blockIdx.y;
  const int b      = bh >> 4;
  const int head   = bh & 15;
  const int qBase  = blockIdx.x * 64 + w * 16;

  const __bf16* qh  = q  + (size_t)bh * Ss * DKk;
  const __bf16* kh  = k  + (size_t)bh * Ss * DKk;
  const __bf16* vth = vt + (size_t)bh * DKk * Ss;

  UB16 qf[2];
  #pragma unroll
  for (int kc = 0; kc < 2; kc++) {
    const __bf16* p = qh + (size_t)(qBase + lane16) * DKk + kc * 32;
    qf[kc].h[0] = *(const v8bf*)(p + hh * 8);
    qf[kc].h[1] = *(const v8bf*)(p + 16 + hh * 8);
  }

  UF8 of[4];
  #pragma unroll
  for (int d = 0; d < 4; d++) {
    v8f z = {0.f,0.f,0.f,0.f,0.f,0.f,0.f,0.f};
    of[d].v = z;
  }
  float mi[8], li[8];
  #pragma unroll
  for (int v = 0; v < 8; v++) { mi[v] = -1e30f; li[v] = 0.f; }

  for (int jb = 0; jb < Ss / 32; jb++) {
    const int keyBase = jb * 32;
    UF8 sc[2];
    #pragma unroll
    for (int jc = 0; jc < 2; jc++) {
      v8f z = {0.f,0.f,0.f,0.f,0.f,0.f,0.f,0.f};
      sc[jc].v = z;
      #pragma unroll
      for (int kc = 0; kc < 2; kc++) {
        UB16 kf;
        const __bf16* p = kh + (size_t)(keyBase + jc * 16 + lane16) * DKk + kc * 32;
        kf.h[0] = *(const v8bf*)(p + hh * 8);
        kf.h[1] = *(const v8bf*)(p + 16 + hh * 8);
        sc[jc].v = wmma_bf16(qf[kc].v, kf.v, sc[jc].v);
      }
    }
    // online softmax over the 32 new columns (rows live across 16-lane halves)
    #pragma unroll
    for (int v = 0; v < 8; v++) {
      float x = fmaxf(sc[0].f[v], sc[1].f[v]);
      x = fmaxf(x, __shfl_xor(x, 1, 32));
      x = fmaxf(x, __shfl_xor(x, 2, 32));
      x = fmaxf(x, __shfl_xor(x, 4, 32));
      x = fmaxf(x, __shfl_xor(x, 8, 32));
      float mn    = fmaxf(mi[v], x);
      float alpha = __expf(mi[v] - mn);
      mi[v] = mn;
      float p0 = __expf(sc[0].f[v] - mn);
      float p1 = __expf(sc[1].f[v] - mn);
      float rs = p0 + p1;
      rs += __shfl_xor(rs, 1, 32);
      rs += __shfl_xor(rs, 2, 32);
      rs += __shfl_xor(rs, 4, 32);
      rs += __shfl_xor(rs, 8, 32);
      li[v] = li[v] * alpha + rs;
      #pragma unroll
      for (int d = 0; d < 4; d++) of[d].f[v] *= alpha;
      int r = v + 8 * hh;                  // C-frag row -> P tile row
      Ps[w][r][lane16]      = (__bf16)p0;
      Ps[w][r][16 + lane16] = (__bf16)p1;
    }
    // re-layout P via per-wave LDS (DS ops are in-order within a wave)
    UB16 pf;
    pf.h[0] = *(const v8bf*)&Ps[w][lane16][hh * 8];
    pf.h[1] = *(const v8bf*)&Ps[w][lane16][16 + hh * 8];
    #pragma unroll
    for (int d = 0; d < 4; d++) {
      UB16 vf;
      const __bf16* p = vth + (size_t)(d * 16 + lane16) * Ss + keyBase;
      vf.h[0] = *(const v8bf*)(p + hh * 8);
      vf.h[1] = *(const v8bf*)(p + 16 + hh * 8);
      of[d].v = wmma_bf16(pf.v, vf.v, of[d].v);
    }
  }

  // normalize + store O in (B,S,D) layout for the out-proj GEMM
  #pragma unroll
  for (int d = 0; d < 4; d++)
    #pragma unroll
    for (int v = 0; v < 8; v++) {
      int row = qBase + v + 8 * hh;
      float val = of[d].f[v] / li[v];
      o[(size_t)(b * Ss + row) * Dd + head * DKk + d * 16 + lane16] = (__bf16)val;
    }
}

// ---------- launcher ----------
extern "C" void kernel_launch(void* const* d_in, const int* in_sizes, int n_in,
                              void* d_out, int out_size, void* d_ws, size_t ws_size,
                              hipStream_t stream) {
  const float* query = (const float*)d_in[0];
  const float* pos   = (const float*)d_in[1];
  const float* wqkv  = (const float*)d_in[2];
  const float* wout  = (const float*)d_in[3];
  float* out = (float*)d_out;

  char*  ws  = (char*)d_ws;
  size_t off = 0;
  auto carve = [&](size_t bytes) -> char* {
    char* p = ws + off;
    off = (off + bytes + 255) & ~(size_t)255;
    return p;
  };
  __bf16* x   = (__bf16*)carve((size_t)Bb * Ss * Dd * 2);        // 8 MB
  __bf16* wq  = (__bf16*)carve((size_t)Dd * 3 * Dd * 2);         // 6 MB
  __bf16* wo  = (__bf16*)carve((size_t)Dd * Dd * 2);             // 2 MB
  __bf16* qb  = (__bf16*)carve((size_t)Bb * Hh * Ss * DKk * 2);  // 8 MB
  __bf16* kb  = (__bf16*)carve((size_t)Bb * Hh * Ss * DKk * 2);  // 8 MB
  __bf16* vtb = (__bf16*)carve((size_t)Bb * Hh * DKk * Ss * 2);  // 8 MB
  __bf16* ob  = (__bf16*)carve((size_t)Bb * Ss * Dd * 2);        // 8 MB

  {
    int total = Bb * Ss * Dd;
    add_pe_cvt<<<total / 4 / 256, 256, 0, stream>>>(query, pos, x, total, Ss * Dd);
  }
  {
    int total = Dd * 3 * Dd;
    cvt_bf16<<<total / 4 / 256, 256, 0, stream>>>(wqkv, wq, total);
  }
  {
    int total = Dd * Dd;
    cvt_bf16<<<total / 4 / 256, 256, 0, stream>>>(wout, wo, total);
  }
  gemm_bf16_128x128<1><<<dim3(3 * Dd / 128, Bb * Ss / 128), 256, 0, stream>>>(
      x, wq, nullptr, Bb * Ss, 3 * Dd, Dd, qb, kb, vtb);
  flash_attn<<<dim3(Ss / 64, Bb * Hh), 128, 0, stream>>>(qb, kb, vtb, ob);
  gemm_bf16_128x128<0><<<dim3(Dd / 128, Bb * Ss / 128), 256, 0, stream>>>(
      ob, wo, out, Bb * Ss, Dd, Dd, nullptr, nullptr, nullptr);
}
